// encoder_30786325578110
// MI455X (gfx1250) — matrix operand order repaired and checked by reading, not compiled, not run on hardware
//
#include <hip/hip_runtime.h>

#define Bsz   256
#define Tsz   1024
#define Hsz   256
#define NFEA  256
#define G4    1024   // 4*H

typedef unsigned int u32;
typedef __attribute__((ext_vector_type(16))) __bf16   v16bf;
typedef __attribute__((ext_vector_type(8)))  __bf16   v8bf;
typedef __attribute__((ext_vector_type(8)))  float    v8f;
typedef __attribute__((ext_vector_type(4)))  u32      v4u;
typedef __attribute__((ext_vector_type(8)))  int      v8i;
typedef __attribute__((ext_vector_type(4)))  int      v4i;

__device__ __forceinline__ v16bf cat8(v8bf lo, v8bf hi) {
  return __builtin_shufflevector(lo, hi, 0,1,2,3,4,5,6,7,8,9,10,11,12,13,14,15);
}

// A fragment (16x32 bf16, MxK) from row-major bf16 (pitch in elements).
// Contiguous k order (same as load_B): lane<16 holds K[0..15]; lane>=16 holds K[16..31].
__device__ __forceinline__ v16bf load_A(const __bf16* base, int pitch, int lane, int k0) {
  const int m   = lane & 15;
  const int sub = (lane >> 4) * 16;
  const __bf16* p = base + m * pitch + k0 + sub;
  v8bf lo = *(const v8bf*)(p);
  v8bf hi = *(const v8bf*)(p + 8);
  return cat8(lo, hi);
}

// B fragment (32x16 bf16, KxN) for D = A @ W^T, W stored row-major [n][k].
// Same contiguous k order as load_A: lane holds column n=lane&15, 16 contiguous K at offset (lane>>4)*16.
__device__ __forceinline__ v16bf load_B(const __bf16* W, int ldk, int lane, int n0, int k0) {
  const int n    = lane & 15;
  const int koff = (lane >> 4) * 16;
  const __bf16* p = W + (size_t)(n0 + n) * ldk + (k0 + koff);
  v8bf lo = *(const v8bf*)(p);
  v8bf hi = *(const v8bf*)(p + 8);
  return cat8(lo, hi);
}

__device__ __forceinline__ float sigmoidf_(float x) { return 1.0f / (1.0f + __expf(-x)); }

// TDM: flat copy of one 32KB xg slab (16384 fp16) into LDS.
// D# per CDNA5 ISA §8.3/8.4: group0 = count|lds_addr|global_addr|type=2,
// group1 = data_size=2B, tensor_dim0=tile_dim0=16384, tensor_dim1=tile_dim1=1.
// This toolchain exposes the 6-arg builtin (extra int32x8 group before cpol).
__device__ __forceinline__ void tdm_load_xg_slab(const _Float16* src, u32 lds_byte_addr) {
  unsigned long long ga = (unsigned long long)(uintptr_t)src;
  v4u g0;
  g0[0] = 1u;                                            // count=1, user descriptor
  g0[1] = lds_byte_addr;                                 // LDS dest (bytes)
  g0[2] = (u32)(ga & 0xFFFFFFFFu);                       // global_addr[31:0]
  g0[3] = (u32)((ga >> 32) & 0x01FFFFFFu) | 0x80000000u; // addr[56:32] | type=2
  v8i g1;
  g1[0] = 0x00010000;   // wg_mask=0, data_size=1 (2 bytes), no pad/iterate
  g1[1] = 0x40000000;   // tensor_dim0[15:0]=16384 in bits 63:48
  g1[2] = 0x00010000;   // tensor_dim0[31:16]=0, tensor_dim1[15:0]=1
  g1[3] = 0x40000000;   // tensor_dim1[31:16]=0, tile_dim0=16384
  g1[4] = 0x00000001;   // tile_dim1=1, tile_dim2=0
  g1[5] = 16384;        // tensor_dim0_stride[31:0]
  g1[6] = 0;            // tensor_dim0_stride[47:32]=0, tensor_dim1_stride[15:0]=0
  g1[7] = 0;
  v4i g2 = {0, 0, 0, 0};                                 // unused (<=2D tensor)
  v4i g3 = {0, 0, 0, 0};                                 // unused (<=2D tensor)
  v8i g4 = {0, 0, 0, 0, 0, 0, 0, 0};                     // unused extra group
  __builtin_amdgcn_tensor_load_to_lds(g0, g1, g2, g3, g4, 0);
}

// ---------------------------------------------------------------- prep ----
__global__ __launch_bounds__(256) void prep_kernel(
    const float* __restrict__ Wih, const float* __restrict__ Whh,
    const float* __restrict__ bih, const float* __restrict__ bhh,
    __bf16* __restrict__ Wih_b, __bf16* __restrict__ Whh_b,
    float* __restrict__ bias) {
  int idx = blockIdx.x * 256 + threadIdx.x;      // covers 1024*256 elements
  Wih_b[idx] = (__bf16)Wih[idx];
  Whh_b[idx] = (__bf16)Whh[idx];
  if (idx < G4) bias[idx] = bih[idx] + bhh[idx];
}

// ----------------------------------------------- big input-projection GEMM
// xg stored fp16, tiled [btile][t][b%16][g] so the recurrent phase reads one
// contiguous 32KB slab per (workgroup, timestep) -> ideal TDM tile.
__global__ __launch_bounds__(256) void xgate_gemm_kernel(
    const float* __restrict__ x, const float* __restrict__ attn,
    const __bf16* __restrict__ Wih_b, const float* __restrict__ bias,
    _Float16* __restrict__ xg) {
  __shared__ __bf16 xa[16][264];                 // 16x256 bf16 A strip (padded)
  const int tid = threadIdx.x;
  const size_t m0 = (size_t)blockIdx.x * 16;
  const int b  = (int)(m0 >> 10);                // T == 1024, strips never cross b
  const int t0 = (int)(m0 & 1023);
  const int r  = tid >> 4;
  const int f0 = tid & 15;
  const float* xrow = x + (m0 + r) * NFEA;
  const float* arow = attn + b * (NFEA / 2);     // N_STOCK = 128
#pragma unroll
  for (int ff = 0; ff < 16; ++ff) {
    int f = f0 + ff * 16;
    xa[r][f] = (__bf16)(xrow[f] * arow[f >> 1]); // repeat(attn,2) fused here
  }
  __syncthreads();

  const int wave = tid >> 5;
  const int lane = tid & 31;
  const int n0   = wave * 128;
  const int mloc = (lane >> 4) * 8;
  const int nn   = lane & 15;
  const int btile = b >> 4;
  const int brow  = b & 15;

  v16bf afr[8];
#pragma unroll
  for (int kk = 0; kk < 8; ++kk) afr[kk] = load_A(&xa[0][0], 264, lane, kk * 32);

#pragma unroll
  for (int nt = 0; nt < 8; ++nt) {
    const int gcol = n0 + nt * 16;
    const float bv = bias[gcol + nn];
    v8f c;
#pragma unroll
    for (int v = 0; v < 8; ++v) c[v] = bv;       // bias folded into C init
#pragma unroll
    for (int kk = 0; kk < 8; ++kk) {
      v16bf bfr = load_B(Wih_b, NFEA, lane, gcol, kk * 32);
      c = __builtin_amdgcn_wmma_f32_16x16x32_bf16(false, afr[kk], false, bfr,
                                                  (short)0, c, false, false);
    }
    // row of D = timestep t0+mloc+v; element -> xg[btile][t][brow][gcol+nn]
    _Float16* op = xg + (((size_t)btile * Tsz + (t0 + mloc)) * 16 + brow) * G4 + gcol + nn;
#pragma unroll
    for (int v = 0; v < 8; ++v) op[(size_t)v * (16 * G4)] = (_Float16)c[v];
  }
}

// ------------------------------------------------------- recurrent scan ---
// One block = one 16-row batch tile. Wave w owns h-columns [32w,32w+32) of
// ALL four gates -> i/f/g/o for (m,j) sit in the same lane: register-only
// cell update. xg slabs are TDM double-buffered into LDS; W_hh streams from
// L2. Two barriers per step.
__global__ __launch_bounds__(256) void lstm_scan_kernel(
    const float* __restrict__ h0, const float* __restrict__ c0,
    const __bf16* __restrict__ Whh_b, const _Float16* __restrict__ xg,
    float* __restrict__ out) {
  __shared__ __bf16   hls[16][264];
  __shared__ _Float16 xls[2][16][G4];            // 2 x 32KB TDM buffers
  const int tid = threadIdx.x;
  const int b0  = blockIdx.x * 16;
  {
    const int r = tid >> 4;
    const int j0 = tid & 15;
#pragma unroll
    for (int jj = 0; jj < 16; ++jj) {
      int j = j0 + jj * 16;
      hls[r][j] = (__bf16)h0[(size_t)(b0 + r) * Hsz + j];
    }
  }
  const int wave  = tid >> 5;
  const int lane  = tid & 31;
  const int mloc  = (lane >> 4) * 8;
  const int nn    = lane & 15;
  const int jbase = wave * 32;

  const _Float16* xg_tile = xg + (size_t)blockIdx.x * Tsz * (16 * G4);
  const u32 xls_lds0 = (u32)(uintptr_t)&xls[0][0][0];
  const u32 xls_lds1 = (u32)(uintptr_t)&xls[1][0][0];

  float cstate[2][8];                            // lane-private c for (m, j)
#pragma unroll
  for (int cb = 0; cb < 2; ++cb)
#pragma unroll
    for (int v = 0; v < 8; ++v)
      cstate[cb][v] = c0[(size_t)(b0 + mloc + v) * Hsz + jbase + cb * 16 + nn];

  if (wave == 0) {                               // prime the 2-deep TDM pipe
    tdm_load_xg_slab(xg_tile, xls_lds0);
    tdm_load_xg_slab(xg_tile + (size_t)16 * G4, xls_lds1);
  }
  __syncthreads();

  for (int t = 0; t < Tsz; ++t) {
    const int cur = t & 1;
    v16bf afr[8];
#pragma unroll
    for (int kk = 0; kk < 8; ++kk) afr[kk] = load_A(&hls[0][0], 264, lane, kk * 32);
    if (wave == 0) __builtin_amdgcn_s_wait_tensorcnt(1);  // slab(t) landed
    __syncthreads();                             // h(t) captured; slab published

    v8f acc[4][2];
#pragma unroll
    for (int q = 0; q < 4; ++q) {                // gate: i, f, g, o
#pragma unroll
      for (int cb = 0; cb < 2; ++cb) {
        const int gcol = q * Hsz + jbase + cb * 16;
        v8f c;
#pragma unroll
        for (int v = 0; v < 8; ++v)              // C init = x-gates from LDS
          c[v] = (float)xls[cur][mloc + v][gcol + nn];
#pragma unroll
        for (int kk = 0; kk < 8; ++kk) {         // K = 256 in 8 WMMA steps
          v16bf bfr = load_B(Whh_b, Hsz, lane, gcol, kk * 32);
          c = __builtin_amdgcn_wmma_f32_16x16x32_bf16(false, afr[kk], false, bfr,
                                                      (short)0, c, false, false);
        }
        acc[q][cb] = c;
      }
    }
    // LSTM cell update: fully lane-local.
#pragma unroll
    for (int cb = 0; cb < 2; ++cb) {
#pragma unroll
      for (int v = 0; v < 8; ++v) {
        const float ig = sigmoidf_(acc[0][cb][v]);
        const float fg = sigmoidf_(acc[1][cb][v]);
        const float gg = tanhf(acc[2][cb][v]);
        const float og = sigmoidf_(acc[3][cb][v]);
        const float cv = fg * cstate[cb][v] + ig * gg;
        cstate[cb][v] = cv;
        const float hv = og * tanhf(cv);
        const int m = mloc + v;
        const int j = jbase + cb * 16 + nn;
        out[((size_t)(b0 + m) * Tsz + t) * Hsz + j] = hv;
        hls[m][j] = (__bf16)hv;                  // h(t+1) for next step's A
      }
    }
    __syncthreads();                             // h(t+1) ready; xls[cur] free
    if (wave == 0 && t + 2 < Tsz)                // refill the freed buffer
      tdm_load_xg_slab(xg_tile + (size_t)(t + 2) * 16 * G4,
                       cur ? xls_lds1 : xls_lds0);
  }
}

// --------------------------------------------------------------- launch ---
extern "C" void kernel_launch(void* const* d_in, const int* in_sizes, int n_in,
                              void* d_out, int out_size, void* d_ws, size_t ws_size,
                              hipStream_t stream) {
  (void)in_sizes; (void)n_in; (void)out_size; (void)ws_size;
  const float* x    = (const float*)d_in[0];
  const float* attn = (const float*)d_in[1];
  const float* h0   = (const float*)d_in[2];
  const float* c0   = (const float*)d_in[3];
  const float* Wih  = (const float*)d_in[4];
  const float* Whh  = (const float*)d_in[5];
  const float* bih  = (const float*)d_in[6];
  const float* bhh  = (const float*)d_in[7];
  float* out = (float*)d_out;

  char* ws = (char*)d_ws;
  __bf16*   Wih_b = (__bf16*)(ws);                               // 512 KB
  __bf16*   Whh_b = (__bf16*)(ws + (size_t)G4 * NFEA * 2);       // 512 KB
  float*    bias  = (float*)(ws + (size_t)G4 * NFEA * 4);        // 4 KB
  _Float16* xg    = (_Float16*)(ws + (size_t)G4 * NFEA * 4 + 8192); // 512 MB

  prep_kernel<<<(G4 * NFEA) / 256, 256, 0, stream>>>(Wih, Whh, bih, bhh,
                                                     Wih_b, Whh_b, bias);
  xgate_gemm_kernel<<<(Bsz * Tsz) / 16, 256, 0, stream>>>(x, attn, Wih_b, bias, xg);
  lstm_scan_kernel<<<Bsz / 16, 256, 0, stream>>>(h0, c0, Whh_b, xg, out);
}
